// VONet_9663676416461
// MI455X (gfx1250) — compile-verified, matrix operand order checked by reading, not run on hardware
//
#include <hip/hip_runtime.h>

#define DIM 384
#define EDGES 20480
#define NPATCH 1280
#define KPAD_CORR 896
#define KIN_CORR 882

typedef __attribute__((ext_vector_type(16))) __bf16 v16bf;
typedef __attribute__((ext_vector_type(8)))  float  v8f;

union FragU { v16bf v; uint4 q[2]; };

__device__ inline v16bf zero_frag() {
  FragU f; f.q[0] = make_uint4(0u,0u,0u,0u); f.q[1] = make_uint4(0u,0u,0u,0u);
  return f.v;
}

// CDNA5 16-bit A/B fragment layout (ISA 7.12.2): lane = {half(1b), idx(4b)}
// elements 0..7  = K[kbase + half*8 .. +7]
// elements 8..15 = K[kbase + 16 + half*8 .. +7]
__device__ inline v16bf load_frag(const __bf16* row, int kbase, int half) {
  FragU f;
  f.q[0] = *reinterpret_cast<const uint4*>(row + kbase + half*8);
  f.q[1] = *reinterpret_cast<const uint4*>(row + kbase + 16 + half*8);
  return f.v;
}

enum { EPI_RELU_BF = 0, EPI_F32 = 1, EPI_ADD_BOTH = 2, EPI_SIG = 3, EPI_GATED = 4 };

// ---------------------------------------------------------------------------
// LDS-staged GEMM, K = 384. One wave = 16-row strip, A hoisted in 96 VGPRs.
// Weight tiles (16x384 bf16 = 12KB, contiguous rows) are DMA'd into LDS with
// global_load_async_to_lds_b128 (ASYNCcnt), double-buffered across the 24
// N-tiles; all 4 waves of the block share each staged tile (4x less B traffic).
// Per-iteration protocol: issue async for tile nt+1 -> compute from LDS tile nt
// (ds_load_b128 fragments -> v_wmma_f32_16x16x32_bf16) -> s_wait_asynccnt 0 ->
// s_barrier (own DMA done + peers done reading the buffer we overwrite next).
// SHIFT = -1/+1: A row remap within aligned 16-blocks (neighbor gather, masked).
// ---------------------------------------------------------------------------
template<int EPI, int SHIFT>
__global__ __launch_bounds__(128)
void gemm_lds(const __bf16* A, const __bf16* W, const float* bias,
              float* o32, __bf16* obf, const float* add1, const float* add2)
{
  constexpr int KT = DIM / 32;                 // 12
  __shared__ __bf16 sb[2][16 * DIM];           // 2 x 12KB double buffer

  const int tid  = threadIdx.x;
  const int wave = tid >> 5;
  const int lane = tid & 31;
  const int half = lane >> 4;
  const int idx  = lane & 15;
  const int mbase = (blockIdx.x * 4 + wave) * 16;

  bool avalid = true;
  int arow;
  if constexpr (SHIFT == 0)      { arow = mbase + idx; }
  else if constexpr (SHIFT < 0)  { avalid = (idx > 0);  arow = mbase + (avalid ? idx - 1 : 0); }
  else                           { avalid = (idx < 15); arow = mbase + (avalid ? idx + 1 : 0); }
  const __bf16* ap = A + (size_t)arow * DIM;

  v16bf afrag[KT];
  #pragma unroll
  for (int kt = 0; kt < KT; ++kt)
    afrag[kt] = avalid ? load_frag(ap, kt * 32, half) : zero_frag();

  // Stage one 12KB weight tile: 128 threads x 6 x b128 lanes-worth of DMA.
  auto stage = [&](int nt, int buf) {
    const uint64_t g0 = (uint64_t)(uintptr_t)(W + (size_t)nt * 16 * DIM) + (uint64_t)tid * 16;
    const uint32_t l0 = (uint32_t)(uintptr_t)(&sb[buf][0]) + (uint32_t)tid * 16;
    #pragma unroll
    for (int i = 0; i < 6; ++i) {
      uint64_t g = g0 + (uint64_t)i * 2048;
      uint32_t l = l0 + (uint32_t)i * 2048;
      asm volatile("global_load_async_to_lds_b128 %0, %1, off"
                   :: "v"(l), "v"(g) : "memory");
    }
  };

  stage(0, 0);
  asm volatile("s_wait_asynccnt 0" ::: "memory");
  __syncthreads();

  for (int nt = 0; nt < DIM / 16; ++nt) {
    const int buf = nt & 1;
    if (nt + 1 < DIM / 16) stage(nt + 1, buf ^ 1);

    const __bf16* wp = &sb[buf][(size_t)idx * DIM];
    v8f acc = {0.f, 0.f, 0.f, 0.f, 0.f, 0.f, 0.f, 0.f};
    #pragma unroll
    for (int kt = 0; kt < KT; ++kt) {
      v16bf b = load_frag(wp, kt * 32, half);      // ds_load_b128 x2
      acc = __builtin_amdgcn_wmma_f32_16x16x32_bf16(false, afrag[kt], false, b,
                                                    (short)0, acc, false, false);
    }

    const int col = nt * 16 + idx;
    const float bv = bias[col];
    #pragma unroll
    for (int r = 0; r < 8; ++r) {
      const int row = mbase + half * 8 + r;
      const size_t off = (size_t)row * DIM + col;
      float v = acc[r] + bv;
      if constexpr (EPI == EPI_RELU_BF) {
        obf[off] = (__bf16)fmaxf(v, 0.f);
      } else if constexpr (EPI == EPI_F32) {
        o32[off] = v;
      } else if constexpr (EPI == EPI_ADD_BOTH) {
        float nv = o32[off] + v;
        o32[off] = nv; obf[off] = (__bf16)nv;
      } else if constexpr (EPI == EPI_SIG) {
        o32[off] = 1.f / (1.f + __expf(-v));
      } else { // EPI_GATED: out = ln_x + gate * res
        o32[off] = add1[off] + add2[off] * v;
      }
    }

    asm volatile("s_wait_asynccnt 0" ::: "memory");
    __syncthreads();
  }
}

// ---------------------------------------------------------------------------
// Direct-load GEMM for the K=896 corr layer (A too large to hoist).
// ---------------------------------------------------------------------------
template<int K, int EPI>
__global__ __launch_bounds__(128)
void gemm_bf16(const __bf16* A, const __bf16* W, const float* bias,
               float* o32, __bf16* obf)
{
  constexpr int KT = K / 32;
  const int wave = threadIdx.x >> 5;
  const int lane = threadIdx.x & 31;
  const int half = lane >> 4;
  const int idx  = lane & 15;
  const int mbase = (blockIdx.x * 4 + wave) * 16;

  const __bf16* ap = A + (size_t)(mbase + idx) * K;

  for (int nt = 0; nt < DIM / 16; ++nt) {
    const __bf16* wp = W + (size_t)(nt * 16 + idx) * K;
    v8f acc = {0.f, 0.f, 0.f, 0.f, 0.f, 0.f, 0.f, 0.f};
    #pragma unroll
    for (int kt = 0; kt < KT; ++kt) {
      v16bf b = load_frag(wp, kt * 32, half);
      v16bf a = load_frag(ap, kt * 32, half);
      acc = __builtin_amdgcn_wmma_f32_16x16x32_bf16(false, a, false, b,
                                                    (short)0, acc, false, false);
    }
    const int col = nt * 16 + idx;
    const float bv = bias[col];
    #pragma unroll
    for (int r = 0; r < 8; ++r) {
      const int row = mbase + half * 8 + r;
      const size_t off = (size_t)row * DIM + col;
      float v = acc[r] + bv;
      if constexpr (EPI == EPI_RELU_BF) obf[off] = (__bf16)fmaxf(v, 0.f);
      else o32[off] = v;
    }
  }
}

enum { LN_RELU_BF = 0, LN_SUM3 = 1, LN_BOTH = 2 };

// Row layernorm (eps=1e-3), one wave per row, 12 features/lane, shuffle reductions.
template<int MODE>
__global__ __launch_bounds__(256)
void ln_rows(const float* a, const float* b, const float* c,
             const float* gamma, const float* beta,
             float* o32, __bf16* obf, int M)
{
  const int wave = threadIdx.x >> 5, lane = threadIdx.x & 31;
  const int row = blockIdx.x * 8 + wave;
  if (row >= M) return;
  const size_t base = (size_t)row * DIM;
  float vals[12];
  float s = 0.f;
  #pragma unroll
  for (int t = 0; t < 12; ++t) {
    const int f = lane + t * 32;
    float v = a[base + f];
    if constexpr (MODE == LN_SUM3) v += b[base + f] + c[base + f];
    vals[t] = v; s += v;
  }
  #pragma unroll
  for (int d = 16; d > 0; d >>= 1) s += __shfl_xor(s, d, 32);
  const float mean = s * (1.f / DIM);
  float q = 0.f;
  #pragma unroll
  for (int t = 0; t < 12; ++t) { float dv = vals[t] - mean; q += dv * dv; }
  #pragma unroll
  for (int d = 16; d > 0; d >>= 1) q += __shfl_xor(q, d, 32);
  const float rs = rsqrtf(q * (1.f / DIM) + 1e-3f);
  #pragma unroll
  for (int t = 0; t < 12; ++t) {
    const int f = lane + t * 32;
    const float y = (vals[t] - mean) * rs * gamma[f] + beta[f];
    if constexpr (MODE == LN_RELU_BF) obf[base + f] = (__bf16)fmaxf(y, 0.f);
    else { o32[base + f] = y; obf[base + f] = (__bf16)y; }
  }
}

// soft_agg over kk: segments are 16 contiguous edges. One wave per segment.
__global__ __launch_bounds__(256)
void red_kk(const float* gx, const float* fx, float* y32, __bf16* ybf)
{
  const int wave = threadIdx.x >> 5, lane = threadIdx.x & 31;
  const int seg = blockIdx.x * 8 + wave;            // 0..NPATCH-1
  const size_t e0 = (size_t)seg * 16;
  for (int t12 = 0; t12 < 12; ++t12) {
    const int f = lane + t12 * 32;
    float g[16];
    float mx = -3.4e38f;
    #pragma unroll
    for (int e = 0; e < 16; ++e) { g[e] = gx[(e0 + e) * DIM + f]; mx = fmaxf(mx, g[e]); }
    float den = 0.f;
    #pragma unroll
    for (int e = 0; e < 16; ++e) { g[e] = __expf(g[e] - mx); den += g[e]; }
    const float inv = 1.f / den;
    float acc = 0.f;
    #pragma unroll
    for (int e = 0; e < 16; ++e) acc += fx[(e0 + e) * DIM + f] * g[e];
    acc *= inv;
    const size_t off = (size_t)seg * DIM + f;
    y32[off] = acc; ybf[off] = (__bf16)acc;
  }
}

// soft_agg over (ii,jj): 256 segments of 80 edges with stride 16. One wave/segment.
__global__ __launch_bounds__(256)
void red_ij(const float* gx, const float* fx, float* y32, __bf16* ybf)
{
  const int wave = threadIdx.x >> 5, lane = threadIdx.x & 31;
  const int seg = blockIdx.x * 8 + wave;            // 0..255
  const int ii = seg >> 4, jj = seg & 15;
  const size_t base = (size_t)ii * 1280 + jj;       // edge id at t=0
  for (int t12 = 0; t12 < 12; ++t12) {
    const int f = lane + t12 * 32;
    float mx = -3.4e38f;
    for (int t = 0; t < 80; ++t)
      mx = fmaxf(mx, gx[(base + (size_t)t * 16) * DIM + f]);
    float den = 0.f;
    for (int t = 0; t < 80; ++t)
      den += __expf(gx[(base + (size_t)t * 16) * DIM + f] - mx);
    const float inv = 1.f / den;
    float acc = 0.f;
    for (int t = 0; t < 80; ++t) {
      const size_t eo = (base + (size_t)t * 16) * DIM + f;
      acc += fx[eo] * __expf(gx[eo] - mx);
    }
    acc *= inv;
    const size_t off = (size_t)seg * DIM + f;
    y32[off] = acc; ybf[off] = (__bf16)acc;
  }
}

// x[e] += hy[seg(e)], refresh bf16 mirror. MODE 0: seg=e>>4 (kk). MODE 1: (ii,jj).
template<int MODE>
__global__ __launch_bounds__(256)
void bcast_add(float* x32, __bf16* xbf, const float* hy)
{
  const size_t i = (size_t)blockIdx.x * 256 + threadIdx.x;
  if (i >= (size_t)EDGES * DIM) return;
  const int e = (int)(i / DIM);
  const int f = (int)(i % DIM);
  int seg;
  if constexpr (MODE == 0) seg = e >> 4;
  else seg = (e / 1280) * 16 + (e & 15);
  const float nv = x32[i] + hy[(size_t)seg * DIM + f];
  x32[i] = nv; xbf[i] = (__bf16)nv;
}

// d = Lin(relu(x)) [E,2]; w = sigmoid(Lin(relu(x))) [E,2]. One wave per edge.
__global__ __launch_bounds__(256)
void heads_kernel(const float* x, const float* dw, const float* db,
                  const float* ww, const float* wb, float* outd, float* outw)
{
  const int wave = threadIdx.x >> 5, lane = threadIdx.x & 31;
  const int e = blockIdx.x * 8 + wave;
  const size_t base = (size_t)e * DIM;
  float s0 = 0.f, s1 = 0.f, s2 = 0.f, s3 = 0.f;
  #pragma unroll
  for (int t = 0; t < 12; ++t) {
    const int f = lane + t * 32;
    const float xv = fmaxf(x[base + f], 0.f);
    s0 += xv * dw[f];       s1 += xv * dw[DIM + f];
    s2 += xv * ww[f];       s3 += xv * ww[DIM + f];
  }
  #pragma unroll
  for (int d = 16; d > 0; d >>= 1) {
    s0 += __shfl_xor(s0, d, 32); s1 += __shfl_xor(s1, d, 32);
    s2 += __shfl_xor(s2, d, 32); s3 += __shfl_xor(s3, d, 32);
  }
  if (lane == 0) {
    outd[2 * e]     = s0 + db[0];
    outd[2 * e + 1] = s1 + db[1];
    outw[2 * e]     = 1.f / (1.f + __expf(-(s2 + wb[0])));
    outw[2 * e + 1] = 1.f / (1.f + __expf(-(s3 + wb[1])));
  }
}

struct WPtrs { const float* p[18]; };

__global__ __launch_bounds__(256)
void conv_wts(WPtrs w, __bf16* dst)
{
  const size_t i = (size_t)blockIdx.x * 256 + threadIdx.x;
  const size_t per = (size_t)DIM * DIM;
  if (i >= 18 * per) return;
  const int m = (int)(i / per);
  const size_t j = i % per;
  dst[i] = (__bf16)w.p[m][j];
}

__global__ __launch_bounds__(256)
void conv_pad(const float* src, __bf16* dst, int rows, int kin, int kout)
{
  const size_t i = (size_t)blockIdx.x * 256 + threadIdx.x;
  if (i >= (size_t)rows * kout) return;
  const int r = (int)(i / kout);
  const int k = (int)(i % kout);
  dst[i] = (__bf16)(k < kin ? src[(size_t)r * kin + k] : 0.f);
}

// ---- input index map (JAX tree-flatten: dict keys sorted at every level) ----
enum {
  IN_CORR = 0, IN_II = 1, IN_INP = 2, IN_JJ = 3, IN_KK = 4, IN_NET = 5,
  IN_AIJ_F_B = 6,  IN_AIJ_F_W = 7,  IN_AIJ_G_B = 8,  IN_AIJ_G_W = 9,
  IN_AIJ_H_B = 10, IN_AIJ_H_W = 11, IN_AKK_F_B = 12, IN_AKK_F_W = 13,
  IN_AKK_G_B = 14, IN_AKK_G_W = 15, IN_AKK_H_B = 16, IN_AKK_H_W = 17,
  IN_C1A_B = 18, IN_C1A_W = 19, IN_C1B_B = 20, IN_C1B_W = 21,
  IN_C2A_B = 22, IN_C2A_W = 23, IN_C2B_B = 24, IN_C2B_W = 25,
  IN_CORR1_B = 26, IN_CORR1_W = 27, IN_CORR2_B = 28, IN_CORR2_W = 29,
  IN_CORR3_B = 30, IN_CORR3_W = 31, IN_CORRLN_B = 32, IN_CORRLN_G = 33,
  IN_D_B = 34, IN_D_W = 35,
  IN_G1G_B = 36, IN_G1G_W = 37, IN_G1R1_B = 38, IN_G1R1_W = 39,
  IN_G1R2_B = 40, IN_G1R2_W = 41, IN_G2G_B = 42, IN_G2G_W = 43,
  IN_G2R1_B = 44, IN_G2R1_W = 45, IN_G2R2_B = 46, IN_G2R2_W = 47,
  IN_GLN1_B = 48, IN_GLN1_G = 49, IN_GLN2_B = 50, IN_GLN2_G = 51,
  IN_NORM_B = 52, IN_NORM_G = 53, IN_W_B = 54, IN_W_W = 55,
};

// bf16 weight slots
enum { W_AIJ_F=0, W_AIJ_G, W_AIJ_H, W_AKK_F, W_AKK_G, W_AKK_H,
       W_C1A, W_C1B, W_C2A, W_C2B, W_CORR2, W_CORR3,
       W_G1G, W_G1R1, W_G1R2, W_G2G, W_G2R1, W_G2R2 };

extern "C" void kernel_launch(void* const* d_in, const int* in_sizes, int n_in,
                              void* d_out, int out_size, void* d_ws, size_t ws_size,
                              hipStream_t stream)
{
  (void)in_sizes; (void)n_in; (void)out_size; (void)ws_size;
  auto F32 = [&](int i) { return (const float*)d_in[i]; };

  char* ws = (char*)d_ws;
  size_t off = 0;
  auto carve = [&](size_t bytes) -> char* {
    char* p = ws + off; off += (bytes + 255) & ~(size_t)255; return p;
  };
  __bf16* wts    = (__bf16*)carve(18ull * DIM * DIM * 2);        // bf16 weights
  __bf16* wc1    = (__bf16*)carve((size_t)DIM * KPAD_CORR * 2);  // corr1.w padded
  __bf16* corrbf = (__bf16*)carve((size_t)EDGES * KPAD_CORR * 2);
  float*  x32    = (float*) carve((size_t)EDGES * DIM * 4);      // master x
  float*  t32    = (float*) carve((size_t)EDGES * DIM * 4);      // f32 temp / gx / gate
  float*  u32    = (float*) carve((size_t)EDGES * DIM * 4);      // fx
  __bf16* xbf    = (__bf16*)carve((size_t)EDGES * DIM * 2);      // bf16 mirror of x
  __bf16* tbf    = (__bf16*)carve((size_t)EDGES * DIM * 2);      // bf16 temp (hidden)
  float*  y32    = (float*) carve((size_t)NPATCH * DIM * 4);
  __bf16* ybf    = (__bf16*)carve((size_t)NPATCH * DIM * 2);
  float*  hy32   = (float*) carve((size_t)NPATCH * DIM * 4);

  float* outx = (float*)d_out;                 // [E,384]
  float* outd = outx + (size_t)EDGES * DIM;    // [E,2]
  float* outw = outd + (size_t)EDGES * 2;      // [E,2]

  auto WB = [&](int i) { return wts + (size_t)i * DIM * DIM; };

  // ---- 0. precision conversion ----
  WPtrs w18;
  const int wsrc[18] = { IN_AIJ_F_W, IN_AIJ_G_W, IN_AIJ_H_W, IN_AKK_F_W, IN_AKK_G_W,
                         IN_AKK_H_W, IN_C1A_W, IN_C1B_W, IN_C2A_W, IN_C2B_W,
                         IN_CORR2_W, IN_CORR3_W, IN_G1G_W, IN_G1R1_W, IN_G1R2_W,
                         IN_G2G_W, IN_G2R1_W, IN_G2R2_W };
  for (int i = 0; i < 18; ++i) w18.p[i] = F32(wsrc[i]);
  conv_wts<<<(18 * DIM * DIM) / 256, 256, 0, stream>>>(w18, wts);
  conv_pad<<<(DIM * KPAD_CORR) / 256, 256, 0, stream>>>(F32(IN_CORR1_W), wc1, DIM, KIN_CORR, KPAD_CORR);
  conv_pad<<<(EDGES * (size_t)KPAD_CORR) / 256, 256, 0, stream>>>(F32(IN_CORR), corrbf, EDGES, KIN_CORR, KPAD_CORR);

  const int GB = EDGES / 64;   // gemm blocks over E rows
  const int LB = EDGES / 8;    // row-wise kernels

  // ---- 1. corr MLP: relu(L1) -> L2 -> LN -> relu -> L3 ----
  gemm_bf16<KPAD_CORR, EPI_RELU_BF><<<GB, 128, 0, stream>>>(
      corrbf, wc1, F32(IN_CORR1_B), nullptr, tbf);
  gemm_lds<EPI_F32, 0><<<GB, 128, 0, stream>>>(
      tbf, WB(W_CORR2), F32(IN_CORR2_B), t32, nullptr, nullptr, nullptr);
  ln_rows<LN_RELU_BF><<<LB, 256, 0, stream>>>(
      t32, nullptr, nullptr, F32(IN_CORRLN_G), F32(IN_CORRLN_B), nullptr, tbf, EDGES);
  gemm_lds<EPI_F32, 0><<<GB, 128, 0, stream>>>(
      tbf, WB(W_CORR3), F32(IN_CORR3_B), t32, nullptr, nullptr, nullptr);

  // ---- 2. x = LN(net + inp + c) ----
  ln_rows<LN_SUM3><<<LB, 256, 0, stream>>>(
      t32, F32(IN_NET), F32(IN_INP), F32(IN_NORM_G), F32(IN_NORM_B), x32, xbf, EDGES);

  // ---- 3. neighbor MLPs (row shift -1 / +1 within 16-blocks, masked) ----
  gemm_lds<EPI_RELU_BF, -1><<<GB, 128, 0, stream>>>(
      xbf, WB(W_C1A), F32(IN_C1A_B), nullptr, tbf, nullptr, nullptr);
  gemm_lds<EPI_ADD_BOTH, 0><<<GB, 128, 0, stream>>>(
      tbf, WB(W_C1B), F32(IN_C1B_B), x32, xbf, nullptr, nullptr);
  gemm_lds<EPI_RELU_BF, 1><<<GB, 128, 0, stream>>>(
      xbf, WB(W_C2A), F32(IN_C2A_B), nullptr, tbf, nullptr, nullptr);
  gemm_lds<EPI_ADD_BOTH, 0><<<GB, 128, 0, stream>>>(
      tbf, WB(W_C2B), F32(IN_C2B_B), x32, xbf, nullptr, nullptr);

  // ---- 4. soft_agg over kk ----
  gemm_lds<EPI_F32, 0><<<GB, 128, 0, stream>>>(
      xbf, WB(W_AKK_G), F32(IN_AKK_G_B), t32, nullptr, nullptr, nullptr);
  gemm_lds<EPI_F32, 0><<<GB, 128, 0, stream>>>(
      xbf, WB(W_AKK_F), F32(IN_AKK_F_B), u32, nullptr, nullptr, nullptr);
  red_kk<<<NPATCH / 8, 256, 0, stream>>>(t32, u32, y32, ybf);
  gemm_lds<EPI_F32, 0><<<NPATCH / 64, 128, 0, stream>>>(
      ybf, WB(W_AKK_H), F32(IN_AKK_H_B), hy32, nullptr, nullptr, nullptr);
  bcast_add<0><<<(EDGES * DIM) / 256, 256, 0, stream>>>(x32, xbf, hy32);

  // ---- 5. soft_agg over (ii,jj) ----
  gemm_lds<EPI_F32, 0><<<GB, 128, 0, stream>>>(
      xbf, WB(W_AIJ_G), F32(IN_AIJ_G_B), t32, nullptr, nullptr, nullptr);
  gemm_lds<EPI_F32, 0><<<GB, 128, 0, stream>>>(
      xbf, WB(W_AIJ_F), F32(IN_AIJ_F_B), u32, nullptr, nullptr, nullptr);
  red_ij<<<256 / 8, 256, 0, stream>>>(t32, u32, y32, ybf);
  gemm_lds<EPI_F32, 0><<<256 / 64, 128, 0, stream>>>(
      ybf, WB(W_AIJ_H), F32(IN_AIJ_H_B), hy32, nullptr, nullptr, nullptr);
  bcast_add<1><<<(EDGES * DIM) / 256, 256, 0, stream>>>(x32, xbf, hy32);

  // ---- 6. gated residual 1: x = LN(x); x = x + sigmoid(Lg x) * (Lr2 relu(Lr1 x)) ----
  ln_rows<LN_BOTH><<<LB, 256, 0, stream>>>(
      x32, nullptr, nullptr, F32(IN_GLN1_G), F32(IN_GLN1_B), x32, xbf, EDGES);
  gemm_lds<EPI_SIG, 0><<<GB, 128, 0, stream>>>(
      xbf, WB(W_G1G), F32(IN_G1G_B), t32, nullptr, nullptr, nullptr);
  gemm_lds<EPI_RELU_BF, 0><<<GB, 128, 0, stream>>>(
      xbf, WB(W_G1R1), F32(IN_G1R1_B), nullptr, tbf, nullptr, nullptr);
  gemm_lds<EPI_GATED, 0><<<GB, 128, 0, stream>>>(
      tbf, WB(W_G1R2), F32(IN_G1R2_B), x32, nullptr, x32, t32);

  // ---- 7. gated residual 2 (result -> d_out x region) ----
  ln_rows<LN_BOTH><<<LB, 256, 0, stream>>>(
      x32, nullptr, nullptr, F32(IN_GLN2_G), F32(IN_GLN2_B), x32, xbf, EDGES);
  gemm_lds<EPI_SIG, 0><<<GB, 128, 0, stream>>>(
      xbf, WB(W_G2G), F32(IN_G2G_B), t32, nullptr, nullptr, nullptr);
  gemm_lds<EPI_RELU_BF, 0><<<GB, 128, 0, stream>>>(
      xbf, WB(W_G2R1), F32(IN_G2R1_B), nullptr, tbf, nullptr, nullptr);
  gemm_lds<EPI_GATED, 0><<<GB, 128, 0, stream>>>(
      tbf, WB(W_G2R2), F32(IN_G2R2_B), outx, nullptr, x32, t32);

  // ---- 8. heads: d, w ----
  heads_kernel<<<LB, 256, 0, stream>>>(
      outx, F32(IN_D_W), F32(IN_D_B), F32(IN_W_W), F32(IN_W_B), outd, outw);
}
